// ImportanceRenderer_82343112999272
// MI455X (gfx1250) — matrix operand order repaired and checked by reading, not compiled, not run on hardware
//
#include <hip/hip_runtime.h>

#define B_  2
#define R_  4096
#define SC  48
#define SF  48
#define CF  32
#define PR  256
#define HID 64
#define ODIM 32
#define NRAY (B_ * R_)                 /* 8192 */
#define NSC  ((long)NRAY * SC)         /* 393216 */
#define NSF  ((long)NRAY * SF)         /* 393216 */
#define RAY_START_F 0.1f
#define RAY_END_F   2.0f

typedef __attribute__((ext_vector_type(16))) _Float16 v16h;
typedef __attribute__((ext_vector_type(8)))  float    v8f;

union AFrag { v16h v; _Float16 e[16]; };

// fast softplus: log(1+exp(x)) via the transcendental pipe (v_exp_f32/v_log_f32)
__device__ __forceinline__ float softplus_f(float x) {
  return (x > 20.f) ? x : __logf(1.f + __expf(x));
}
__device__ __forceinline__ float sigmoid_f(float x) {
  return 1.f / (1.f + __expf(-x));
}

// grid_sample2d semantics from the reference (zero padding, bilinear)
__device__ __forceinline__ float sample_plane(const float* __restrict__ f, float gx, float gy) {
  float x = ((gx + 1.f) * (float)PR - 1.f) * 0.5f;
  float y = ((gy + 1.f) * (float)PR - 1.f) * 0.5f;
  float x0f = floorf(x), y0f = floorf(y);
  int x0 = (int)x0f, y0 = (int)y0f;
  float wx1 = x - x0f, wy1 = y - y0f;
  float wx0 = 1.f - wx1, wy0 = 1.f - wy1;
  float acc = 0.f;
#pragma unroll
  for (int dy = 0; dy < 2; ++dy) {
    int yi = y0 + dy;
    float wy = dy ? wy1 : wy0;
    bool yin = (yi >= 0) && (yi < PR);
    int yc = yi < 0 ? 0 : (yi > PR - 1 ? PR - 1 : yi);
#pragma unroll
    for (int dx = 0; dx < 2; ++dx) {
      int xi = x0 + dx;
      float wx = dx ? wx1 : wx0;
      bool xin = (xi >= 0) && (xi < PR);
      int xc = xi < 0 ? 0 : (xi > PR - 1 ? PR - 1 : xi);
      float v = f[yc * PR + xc];
      acc += ((yin && xin) ? v : 0.f) * (wy * wx);
    }
  }
  return acc;
}

// ---------------------------------------------------------------------------
// Kernel 1: coarse stratified depths + WMMA weight pre-pack.
// B-matrix (16-bit, 32x16) pre-pack k order used here (same as A): K = t + 16*(lane>=16), N = lane&15.
// ---------------------------------------------------------------------------
__global__ __launch_bounds__(256) void prep_kernel(
    const float* __restrict__ noise_strat,
    const float* __restrict__ w1,
    const float* __restrict__ w2,
    float* __restrict__ depths_c,
    _Float16* __restrict__ w1p,
    _Float16* __restrict__ w2p)
{
  long i = (long)blockIdx.x * 256 + threadIdx.x;
  if (i < NSC) {
    int s = (int)(i % SC);
    float delta = (RAY_END_F - RAY_START_F) / (float)(SC - 1);
    float base  = RAY_START_F + delta * (float)s;
    depths_c[i] = base + noise_strat[i] * delta;
  }
  if (i < 2048) {                       // w1: 4 ntiles * 32 lanes * 16 halves
    int nt = (int)(i >> 9);
    int lane = (int)((i >> 4) & 31);
    int t = (int)(i & 15);
    int K = t + ((lane >= 16) ? 16 : 0);
    int N = nt * 16 + (lane & 15);
    w1p[i] = (_Float16)w1[K * HID + N];
  }
  if (i < 3072) {                       // w2: 2 kchunks * 3 ntiles * 32 * 16
    int kc = (i >= 1536) ? 1 : 0;
    long r = i - (long)kc * 1536;
    int nt = (int)(r >> 9);
    int lane = (int)((r >> 4) & 31);
    int t = (int)(r & 15);
    int K = kc * 32 + t + ((lane >= 16) ? 16 : 0);
    int N = nt * 16 + (lane & 15);
    w2p[i] = (N < 1 + ODIM) ? (_Float16)w2[K * (1 + ODIM) + N] : (_Float16)0.f;
  }
}

// ---------------------------------------------------------------------------
// Kernel 2: triplane gather + 2-layer MLP via v_wmma_f32_16x16x32_f16.
// One wave = one tile of 16 samples; lane = plane channel for the gather.
// ---------------------------------------------------------------------------
__global__ __launch_bounds__(256) void model_kernel(
    const float* __restrict__ planes,
    const float* __restrict__ orig,
    const float* __restrict__ dirs,
    const float* __restrict__ depths,
    const _Float16* __restrict__ w1p,
    const _Float16* __restrict__ w2p,
    const float* __restrict__ b1,
    const float* __restrict__ b2,
    float* __restrict__ sigma_out,
    _Float16* __restrict__ rgb_out,
    int S)
{
  __shared__ _Float16 lds[8][16 * CF + 16 * HID];   // per-wave: feat(512) + h(1024)
  const int wave = (int)(threadIdx.x >> 5);
  const int lane = (int)(threadIdx.x & 31);
  _Float16* featL = lds[wave];
  _Float16* hL    = lds[wave] + 16 * CF;

  const long tile = (long)blockIdx.x * 8 + wave;
  const long base = tile * 16;

  // ---- triplane feature gather (lane == channel, 32 channels) ----
  {
    const int c = lane;
#pragma unroll 1
    for (int m = 0; m < 16; ++m) {
      long sid = base + m;
      int  s   = (int)(sid % S);
      long br  = sid / S;
      float t  = depths[br * S + s];
      float ox = orig[br * 3 + 0], oy = orig[br * 3 + 1], oz = orig[br * 3 + 2];
      float dx = dirs[br * 3 + 0], dy = dirs[br * 3 + 1], dz = dirs[br * 3 + 2];
      float cx = (ox + t * dx) * 0.5f;     // 2 / BOX_WARP = 0.5
      float cy = (oy + t * dy) * 0.5f;
      float cz = (oz + t * dz) * 0.5f;
      const float* pb = planes + (br / R_) * (3L * CF * PR * PR) + (long)c * PR * PR;
      float acc = sample_plane(pb + 0L * CF * PR * PR, cx, cy)
                + sample_plane(pb + 1L * CF * PR * PR, cy, cz)
                + sample_plane(pb + 2L * CF * PR * PR, cz, cx);
      featL[m * CF + c] = (_Float16)(acc * (1.f / 3.f));
    }
  }
  __syncthreads();

  // A-matrix (16-bit, 16x32) k order used here (matches w1p/w2p packing):
  //   M = lane&15, K = t + 16*(lane>=16)
  const int M    = lane & 15;
  const int kofs = (lane >> 4) << 3;     // 0 or 8

  AFrag a1;
#pragma unroll
  for (int t = 0; t < 16; ++t) {
    int K = t + ((lane >> 4) << 4);
    a1.e[t] = featL[M * CF + K];
  }

  // ---- layer 1: 16x64 = 4 WMMA n-tiles ----
  v8f h[4];
#pragma unroll
  for (int nt = 0; nt < 4; ++nt) {
    v16h bm = *(const v16h*)(w1p + ((long)nt * 32 + lane) * 16);
    v8f z = {0.f, 0.f, 0.f, 0.f, 0.f, 0.f, 0.f, 0.f};
    h[nt] = __builtin_amdgcn_wmma_f32_16x16x32_f16(false, a1.v, false, bm,
                                                   (short)0, z, false, false);
  }
  // bias + softplus, stage h into LDS (C layout: m = g + 8*(lane>=16), n = nt*16 + lane&15)
#pragma unroll
  for (int nt = 0; nt < 4; ++nt) {
    int n = nt * 16 + (lane & 15);
    float bias = b1[n];
#pragma unroll
    for (int g = 0; g < 8; ++g) {
      int m = g + kofs;
      hL[m * HID + n] = (_Float16)softplus_f(h[nt][g] + bias);
    }
  }
  __syncthreads();

  // ---- layer 2: K=64 (two k-chunks), N=33 padded to 48 (3 n-tiles) ----
  AFrag a2[2];
#pragma unroll
  for (int kc = 0; kc < 2; ++kc) {
#pragma unroll
    for (int t = 0; t < 16; ++t) {
      int K = t + ((lane >> 4) << 4);
      a2[kc].e[t] = hL[M * HID + kc * 32 + K];
    }
  }

#pragma unroll
  for (int nt = 0; nt < 3; ++nt) {
    v8f acc = {0.f, 0.f, 0.f, 0.f, 0.f, 0.f, 0.f, 0.f};
    v16h b0 = *(const v16h*)(w2p + (((long)0 * 3 + nt) * 32 + lane) * 16);
    acc = __builtin_amdgcn_wmma_f32_16x16x32_f16(false, a2[0].v, false, b0,
                                                 (short)0, acc, false, false);
    v16h bB = *(const v16h*)(w2p + (((long)1 * 3 + nt) * 32 + lane) * 16);
    acc = __builtin_amdgcn_wmma_f32_16x16x32_f16(false, a2[1].v, false, bB,
                                                 (short)0, acc, false, false);
    int n = nt * 16 + (lane & 15);
    if (n < 1 + ODIM) {
      float bias = b2[n];
#pragma unroll
      for (int g = 0; g < 8; ++g) {
        int m = g + kofs;
        long sid = base + m;
        float o = acc[g] + bias;
        if (n == 0) {
          sigma_out[sid] = o;                                  // raw sigma
        } else {
          float v = sigmoid_f(o) * 1.002f - 0.001f;            // sigmoid*(1+2e-3)-1e-3
          rgb_out[sid * ODIM + (n - 1)] = (_Float16)v;
        }
      }
    }
  }
}

// ---------------------------------------------------------------------------
// Kernel 3: coarse ray-march weights + importance (inverse-CDF) sampling.
// One thread per ray.
// ---------------------------------------------------------------------------
__global__ __launch_bounds__(256) void march_imp_kernel(
    const float* __restrict__ depths_c,
    const float* __restrict__ sigma_c,
    const float* __restrict__ noise_imp,
    float* __restrict__ depths_f)
{
  int ray = blockIdx.x * blockDim.x + threadIdx.x;
  if (ray >= NRAY) return;
  const float* d  = depths_c + (long)ray * SC;
  const float* sg = sigma_c  + (long)ray * SC;

  float w[SC - 1];
  float T = 1.f;
#pragma unroll 1
  for (int i = 0; i < SC - 1; ++i) {
    float dm    = 0.5f * (sg[i] + sg[i + 1]) - 1.f;
    float sp    = softplus_f(dm);
    float delta = d[i + 1] - d[i];
    float alpha = 1.f - __expf(-sp * delta);
    w[i] = alpha * T;
    T *= (1.f - alpha + 1e-10f);
  }
  // max-filter + blur: wp = pad(w, -inf); mp[i]=max(wp[i],wp[i+1]); wf = 0.5*(mp[i]+mp[i+1])+0.01
  float mp[SC];
  mp[0] = w[0];
#pragma unroll 1
  for (int i = 1; i < SC - 1; ++i) mp[i] = fmaxf(w[i - 1], w[i]);
  mp[SC - 1] = w[SC - 2];
  float wf[SC - 1];
#pragma unroll 1
  for (int i = 0; i < SC - 1; ++i) wf[i] = 0.5f * (mp[i] + mp[i + 1]) + 0.01f;
  float bins[SC - 1];
#pragma unroll 1
  for (int i = 0; i < SC - 1; ++i) bins[i] = 0.5f * (d[i] + d[i + 1]);

  // sample_pdf over wf[1..45] (45 weights) -> cdf[0..45]
  float cdf[SC - 2];
  float tot = 0.f;
#pragma unroll 1
  for (int i = 0; i < 45; ++i) tot += wf[i + 1] + 1e-5f;
  cdf[0] = 0.f;
  float run = 0.f;
#pragma unroll 1
  for (int i = 0; i < 45; ++i) { run += (wf[i + 1] + 1e-5f) / tot; cdf[i + 1] = run; }

  const float* u  = noise_imp + (long)ray * SF;
  float* df       = depths_f  + (long)ray * SF;
#pragma unroll 1
  for (int j = 0; j < SF; ++j) {
    float uu = u[j];
    int ind = 0;
#pragma unroll 1
    for (int i = 0; i < 46; ++i) ind += (cdf[i] <= uu) ? 1 : 0;   // searchsorted right
    int below = ind - 1 > 0 ? ind - 1 : 0;
    int above = ind < 45 ? ind : 45;
    float cb = cdf[below], ca = cdf[above];
    float bb = bins[below], ba = bins[above];
    float den = (ca - cb < 1e-5f) ? 1.f : (ca - cb);
    df[j] = bb + (uu - cb) / den * (ba - bb);
  }
}

// ---------------------------------------------------------------------------
// Kernel 4: unify (sort+merge) + final ray-march. One thread per ray.
// Writes depth / weight-sum / visibility into d_out; weights + order to ws.
// ---------------------------------------------------------------------------
__global__ __launch_bounds__(256) void unify_kernel(
    const float* __restrict__ depths_c, const float* __restrict__ sigma_c,
    const float* __restrict__ depths_f, const float* __restrict__ sigma_f,
    float* __restrict__ out, float* __restrict__ wts, int* __restrict__ ord)
{
  int ray = blockIdx.x * blockDim.x + threadIdx.x;
  if (ray >= NRAY) return;

  float dfv[SF]; int fidx[SF];
#pragma unroll 1
  for (int j = 0; j < SF; ++j) { dfv[j] = depths_f[(long)ray * SF + j]; fidx[j] = j; }
#pragma unroll 1
  for (int j = 1; j < SF; ++j) {               // insertion sort (fine depths unsorted)
    float dv = dfv[j]; int iv = fidx[j];
    int k = j - 1;
    while (k >= 0 && dfv[k] > dv) { dfv[k + 1] = dfv[k]; fidx[k + 1] = fidx[k]; --k; }
    dfv[k + 1] = dv; fidx[k + 1] = iv;
  }

  const int NT = SC + SF;                      // 96
  float D[SC + SF]; float Sg[SC + SF];
  int i = 0, j = 0;
#pragma unroll 1
  for (int k = 0; k < NT; ++k) {               // merge (coarse already sorted)
    float dc = (i < SC) ? depths_c[(long)ray * SC + i] : 3.4e38f;
    float dfx = (j < SF) ? dfv[j] : 3.4e38f;
    int o;
    if (dc <= dfx) {
      D[k] = dc; Sg[k] = sigma_c[(long)ray * SC + i]; o = i; ++i;
    } else {
      D[k] = dfx; Sg[k] = sigma_f[(long)ray * SF + fidx[j]]; o = SC + fidx[j]; ++j;
    }
    ord[(long)ray * NT + k] = o;
  }

  float T = 1.f, wtot = 0.f, dsum = 0.f;
#pragma unroll 1
  for (int k = 0; k < NT - 1; ++k) {
    float dm    = 0.5f * (Sg[k] + Sg[k + 1]) - 1.f;
    float sp    = softplus_f(dm);
    float delta = D[k + 1] - D[k];
    float alpha = 1.f - __expf(-sp * delta);
    float wgt   = alpha * T;
    wts[(long)ray * (NT - 1) + k] = wgt;
    wtot += wgt;
    dsum += wgt * 0.5f * (D[k] + D[k + 1]);
    T *= (1.f - alpha + 1e-10f);
  }
  float depth = dsum / wtot;
  if (!(depth == depth)) depth = 3.4e38f;      // nan -> inf
  depth = fminf(fmaxf(depth, D[0]), D[NT - 1]);

  out[(long)NRAY * ODIM + ray]        = depth;   // depth_final
  out[(long)NRAY * (ODIM + 1) + ray]  = wtot;    // weights.sum
  out[(long)NRAY * (ODIM + 2) + ray]  = T;       // visibility
}

// ---------------------------------------------------------------------------
// Kernel 5: weighted color accumulation. One thread per (ray, channel).
// ---------------------------------------------------------------------------
__global__ __launch_bounds__(256) void rgb_kernel(
    const _Float16* __restrict__ rgb_c, const _Float16* __restrict__ rgb_f,
    const float* __restrict__ wts, const int* __restrict__ ord,
    float* __restrict__ out)
{
  long idx = (long)blockIdx.x * blockDim.x + threadIdx.x;
  int  c   = (int)(idx & (ODIM - 1));
  long ray = idx >> 5;
  if (ray >= NRAY) return;
  const int*   O = ord + ray * (SC + SF);
  const float* W = wts + ray * (SC + SF - 1);

  int o0 = O[0];
  float prev = (o0 < SC) ? (float)rgb_c[(ray * SC + o0) * ODIM + c]
                         : (float)rgb_f[(ray * SF + (o0 - SC)) * ODIM + c];
  float acc = 0.f;
#pragma unroll 1
  for (int k = 0; k < SC + SF - 1; ++k) {
    int o = O[k + 1];
    float nxt = (o < SC) ? (float)rgb_c[(ray * SC + o) * ODIM + c]
                         : (float)rgb_f[(ray * SF + (o - SC)) * ODIM + c];
    acc += W[k] * 0.5f * (prev + nxt);
    prev = nxt;
  }
  out[ray * ODIM + c] = acc * 2.f - 1.f;
}

// ---------------------------------------------------------------------------
extern "C" void kernel_launch(void* const* d_in, const int* in_sizes, int n_in,
                              void* d_out, int out_size, void* d_ws, size_t ws_size,
                              hipStream_t stream)
{
  (void)in_sizes; (void)n_in; (void)out_size; (void)ws_size;
  const float* planes = (const float*)d_in[0];
  const float* orig   = (const float*)d_in[1];
  const float* dirs   = (const float*)d_in[2];
  const float* w1     = (const float*)d_in[3];
  const float* b1     = (const float*)d_in[4];
  const float* w2     = (const float*)d_in[5];
  const float* b2     = (const float*)d_in[6];
  const float* nstrat = (const float*)d_in[7];
  const float* nimp   = (const float*)d_in[8];
  float* out = (float*)d_out;

  char* ws = (char*)d_ws;
  size_t off = 0;
  auto alloc = [&](size_t bytes) {
    void* p = ws + off;
    off += (bytes + 255) & ~(size_t)255;
    return p;
  };
  float*    depths_c = (float*)alloc((size_t)NSC * 4);
  float*    depths_f = (float*)alloc((size_t)NSF * 4);
  float*    sigma_c  = (float*)alloc((size_t)NSC * 4);
  float*    sigma_f  = (float*)alloc((size_t)NSF * 4);
  _Float16* rgb_c    = (_Float16*)alloc((size_t)NSC * ODIM * 2);
  _Float16* rgb_f    = (_Float16*)alloc((size_t)NSF * ODIM * 2);
  float*    wtsb     = (float*)alloc((size_t)NRAY * (SC + SF - 1) * 4);
  int*      ordb     = (int*)alloc((size_t)NRAY * (SC + SF) * 4);
  _Float16* w1p      = (_Float16*)alloc(2048 * 2);
  _Float16* w2p      = (_Float16*)alloc(3072 * 2);

  prep_kernel<<<(int)(NSC / 256), 256, 0, stream>>>(nstrat, w1, w2, depths_c, w1p, w2p);
  model_kernel<<<(int)(NSC / 128), 256, 0, stream>>>(planes, orig, dirs, depths_c,
                                                     w1p, w2p, b1, b2, sigma_c, rgb_c, SC);
  march_imp_kernel<<<NRAY / 256, 256, 0, stream>>>(depths_c, sigma_c, nimp, depths_f);
  model_kernel<<<(int)(NSF / 128), 256, 0, stream>>>(planes, orig, dirs, depths_f,
                                                     w1p, w2p, b1, b2, sigma_f, rgb_f, SF);
  unify_kernel<<<NRAY / 256, 256, 0, stream>>>(depths_c, sigma_c, depths_f, sigma_f,
                                               out, wtsb, ordb);
  rgb_kernel<<<(NRAY * ODIM) / 256, 256, 0, stream>>>(rgb_c, rgb_f, wtsb, ordb, out);
}
